// AudioLSTM_69080253989469
// MI455X (gfx1250) — matrix operand order repaired and checked by reading, not compiled, not run on hardware
//
#include <hip/hip_runtime.h>

// ---------------------------------------------------------------------------
// Problem constants (match the reference)
// ---------------------------------------------------------------------------
#define NUM_LAYERS 3
#define HID 1024          // H
#define FEAT 2048         // IN == 2*H, every layer input width
#define TSTEPS 4096       // T
#define G4H (4 * HID)     // 4096 gate rows

typedef __attribute__((ext_vector_type(16))) __bf16 v16bf;
typedef __attribute__((ext_vector_type(8)))  float  v8f;

struct __attribute__((aligned(16))) U128 { unsigned int a, b, c, d; };
union Frag { U128 q[2]; v16bf v; };

// ---------------------------------------------------------------------------
// Helpers
// ---------------------------------------------------------------------------
__device__ __forceinline__ unsigned short f2bf_rne(float f) {
    unsigned int u = __float_as_uint(f);
    unsigned int r = u + 0x7FFFu + ((u >> 16) & 1u);
    return (unsigned short)(r >> 16);
}
__device__ __forceinline__ float bflo(unsigned int p) {
    return __uint_as_float((p & 0xFFFFu) << 16);
}
__device__ __forceinline__ float bfhi(unsigned int p) {
    return __uint_as_float(p & 0xFFFF0000u);
}
__device__ __forceinline__ float sigm(float x) {
    return 1.0f / (1.0f + __expf(-x));
}

// ---------------------------------------------------------------------------
// fp32 -> bf16 (RNE), 4 elements / thread
// ---------------------------------------------------------------------------
__global__ void f32_to_bf16_vec(const float* __restrict__ in,
                                unsigned int* __restrict__ out2, int n4) {
    int i = blockIdx.x * blockDim.x + threadIdx.x;
    if (i < n4) {
        float4 v = ((const float4*)in)[i];
        unsigned int lo = (unsigned int)f2bf_rne(v.x) | ((unsigned int)f2bf_rne(v.y) << 16);
        unsigned int hi = (unsigned int)f2bf_rne(v.z) | ((unsigned int)f2bf_rne(v.w) << 16);
        out2[2 * i]     = lo;
        out2[2 * i + 1] = hi;
    }
}

// ---------------------------------------------------------------------------
// Zero the grid-barrier state (run once per kernel_launch)
// ---------------------------------------------------------------------------
__global__ void zero_bar(unsigned int* bar) {
    if (threadIdx.x < 32) bar[threadIdx.x] = 0u;
}

// ---------------------------------------------------------------------------
// Fragment loaders (ISA 16-bit WMMA layouts)
// ---------------------------------------------------------------------------
__device__ __forceinline__ v16bf load_fragA(const unsigned short* row, int kb, int lane) {
    Frag f;
    int koff = (lane >> 4) * 8;                        // A 16x32: two 16B chunks
    f.q[0] = *(const U128*)(row + kb + koff);          // K = koff .. koff+7
    f.q[1] = *(const U128*)(row + kb + koff + 16);     // K = koff+16 .. koff+23
    return f.v;
}
__device__ __forceinline__ v16bf load_fragB(const unsigned short* row, int kb, int lane) {
    Frag f;
    int koff = (lane >> 4) * 8;                        // B 32x16: two 16B chunks
    f.q[0] = *(const U128*)(row + kb + koff);
    f.q[1] = *(const U128*)(row + kb + koff + 16);
    return f.v;
}

// ---------------------------------------------------------------------------
// WMMA GEMM: C(T x 4096) = A(T x 2048, bf16) * W^T (W: 4096 x 2048, bf16)
//            + (b_ih + b_hh) per column.
// Block tile 128x64: 8 waves, each wave 16 rows x 64 cols (4 wmma tiles).
// Software-pipelined: next K-step fragments load while current WMMAs run.
// ---------------------------------------------------------------------------
__global__ __launch_bounds__(256) void gemm_preact(
    const unsigned short* __restrict__ A,   // T x 2048 bf16
    const unsigned short* __restrict__ W,   // 4096 x 2048 bf16
    const float* __restrict__ bih,          // 4096
    const float* __restrict__ bhh,          // 4096
    float* __restrict__ C) {                // T x 4096 f32
    const int K = FEAT, N = G4H;
    const int wave = threadIdx.x >> 5;
    const int lane = threadIdx.x & 31;

    const int m0 = blockIdx.x * 128 + wave * 16;
    const int n0 = blockIdx.y * 64;

    const unsigned short* arow = A + (size_t)(m0 + (lane & 15)) * K;
    const unsigned short* brow = W + (size_t)(n0 + (lane & 15)) * K;

    v8f acc0 = {}, acc1 = {}, acc2 = {}, acc3 = {};

    // prologue: fragments for kb = 0
    v16bf a_c  = load_fragA(arow, 0, lane);
    v16bf b0_c = load_fragB(brow,                    0, lane);
    v16bf b1_c = load_fragB(brow + 16 * (size_t)K,   0, lane);
    v16bf b2_c = load_fragB(brow + 32 * (size_t)K,   0, lane);
    v16bf b3_c = load_fragB(brow + 48 * (size_t)K,   0, lane);

    for (int kb = 32; kb < K; kb += 32) {
        // issue next iteration's loads first (overlap with the 4 WMMAs below)
        v16bf a_n  = load_fragA(arow, kb, lane);
        v16bf b0_n = load_fragB(brow,                  kb, lane);
        v16bf b1_n = load_fragB(brow + 16 * (size_t)K, kb, lane);
        v16bf b2_n = load_fragB(brow + 32 * (size_t)K, kb, lane);
        v16bf b3_n = load_fragB(brow + 48 * (size_t)K, kb, lane);
        if (kb + 96 < K) {
            __builtin_prefetch(arow + kb + 96, 0, 1);
            __builtin_prefetch(brow + kb + 96, 0, 1);
        }
        acc0 = __builtin_amdgcn_wmma_f32_16x16x32_bf16(false, a_c, false, b0_c, (short)0, acc0, false, false);
        acc1 = __builtin_amdgcn_wmma_f32_16x16x32_bf16(false, a_c, false, b1_c, (short)0, acc1, false, false);
        acc2 = __builtin_amdgcn_wmma_f32_16x16x32_bf16(false, a_c, false, b2_c, (short)0, acc2, false, false);
        acc3 = __builtin_amdgcn_wmma_f32_16x16x32_bf16(false, a_c, false, b3_c, (short)0, acc3, false, false);
        a_c = a_n; b0_c = b0_n; b1_c = b1_n; b2_c = b2_n; b3_c = b3_n;
    }
    // epilogue
    acc0 = __builtin_amdgcn_wmma_f32_16x16x32_bf16(false, a_c, false, b0_c, (short)0, acc0, false, false);
    acc1 = __builtin_amdgcn_wmma_f32_16x16x32_bf16(false, a_c, false, b1_c, (short)0, acc1, false, false);
    acc2 = __builtin_amdgcn_wmma_f32_16x16x32_bf16(false, a_c, false, b2_c, (short)0, acc2, false, false);
    acc3 = __builtin_amdgcn_wmma_f32_16x16x32_bf16(false, a_c, false, b3_c, (short)0, acc3, false, false);

    // D layout: lanes 0-15 -> N=lane, M=r ; lanes 16-31 -> N=lane-16, M=r+8
    const int rbase = m0 + ((lane >> 4) << 3);
    const int cbase = n0 + (lane & 15);
    v8f accs[4] = {acc0, acc1, acc2, acc3};
#pragma unroll
    for (int t = 0; t < 4; ++t) {
        int col = cbase + t * 16;
        float bias = bih[col] + bhh[col];
#pragma unroll
        for (int r = 0; r < 8; ++r) {
            C[(size_t)(rbase + r) * N + col] = accs[t][r] + bias;
        }
    }
}

// ---------------------------------------------------------------------------
// Device-wide sense barrier over `nblocks` co-scheduled workgroups
// ---------------------------------------------------------------------------
__device__ __forceinline__ void grid_barrier(unsigned int* bar, int nblocks) {
    __syncthreads();
    if (threadIdx.x == 0) {
        unsigned int my = __hip_atomic_load(&bar[1], __ATOMIC_ACQUIRE, __HIP_MEMORY_SCOPE_AGENT);
        unsigned int arr = __hip_atomic_fetch_add(&bar[0], 1u, __ATOMIC_ACQ_REL, __HIP_MEMORY_SCOPE_AGENT);
        if (arr == (unsigned int)nblocks - 1u) {
            __hip_atomic_store(&bar[0], 0u, __ATOMIC_RELAXED, __HIP_MEMORY_SCOPE_AGENT);
            __hip_atomic_fetch_add(&bar[1], 1u, __ATOMIC_RELEASE, __HIP_MEMORY_SCOPE_AGENT);
        } else {
            while (__hip_atomic_load(&bar[1], __ATOMIC_ACQUIRE, __HIP_MEMORY_SCOPE_AGENT) == my)
                __builtin_amdgcn_s_sleep(1);
        }
    }
    __syncthreads();
}

// ---------------------------------------------------------------------------
// Fused bidirectional persistent recurrence for one layer.
// 128 blocks x 256 threads: blocks [0,64) = forward, [64,128) = backward,
// each direction has its own barrier counters, h-state and pre-activations.
// Per block: 16 hidden units; thread = (unit, gate, k-slice of 256).
// ---------------------------------------------------------------------------
__global__ __launch_bounds__(256) void lstm_recur_bidir(
    const float* __restrict__ pre0,          // T x 4096 fwd pre-activations
    const float* __restrict__ pre1,          // T x 4096 bwd pre-activations
    const unsigned short* __restrict__ whh0, // 4096 x 1024 bf16 fwd
    const unsigned short* __restrict__ whh1, // 4096 x 1024 bf16 bwd
    float* __restrict__ h0g, float* __restrict__ h1g,   // 1024 each
    float* __restrict__ seq_out,             // T x 2048
    float* __restrict__ hn0,                 // -> h_n[2l+0], +HID = [2l+1]
    float* __restrict__ cn0,
    unsigned int* __restrict__ bar) {        // bar[0..] dir0, bar[16..] dir1
    __shared__ float hs[HID];
    __shared__ float pbuf[256];
    __shared__ float gbuf[16 * 4];

    const int dgrp = blockIdx.x >> 6;        // 0 fwd, 1 bwd (= reverse flag)
    const int blk  = blockIdx.x & 63;
    const int tid  = threadIdx.x;
    const int u    = tid >> 4;               // 0..15 unit-in-block
    const int gate = (tid >> 2) & 3;         // 0..3
    const int ks   = tid & 3;                // 0..3 k-slice
    const int j    = blk * 16 + u;           // hidden unit 0..1023

    const float* pre            = dgrp ? pre1 : pre0;
    const unsigned short* whh   = dgrp ? whh1 : whh0;
    float* h_state              = dgrp ? h1g : h0g;
    unsigned int* mybar         = bar + dgrp * 16;
    const unsigned short* wseg  = whh + (size_t)(gate * HID + j) * HID + ks * 256;

    if (blk == 0) {
        for (int k = tid; k < HID; k += 256) h_state[k] = 0.0f;
    }
    __threadfence();
    grid_barrier(mybar, 64);

    float creg = 0.0f, hreg = 0.0f;

    for (int s = 0; s < TSTEPS; ++s) {
        const int t = dgrp ? (TSTEPS - 1 - s) : s;

        for (int k = tid; k < HID; k += 256) hs[k] = h_state[k];
        __syncthreads();

        const float* hp = hs + ks * 256;
        float acc = 0.0f;
#pragma unroll 4
        for (int kk = 0; kk < 256; kk += 8) {
            U128 q = *(const U128*)(wseg + kk);
            acc = fmaf(hp[kk + 0], bflo(q.a), acc);
            acc = fmaf(hp[kk + 1], bfhi(q.a), acc);
            acc = fmaf(hp[kk + 2], bflo(q.b), acc);
            acc = fmaf(hp[kk + 3], bfhi(q.b), acc);
            acc = fmaf(hp[kk + 4], bflo(q.c), acc);
            acc = fmaf(hp[kk + 5], bfhi(q.c), acc);
            acc = fmaf(hp[kk + 6], bflo(q.d), acc);
            acc = fmaf(hp[kk + 7], bfhi(q.d), acc);
        }
        pbuf[tid] = acc;
        __syncthreads();

        if (ks == 0) {  // combine 4 k-slices + precomputed input contribution
            float g = pbuf[tid] + pbuf[tid + 1] + pbuf[tid + 2] + pbuf[tid + 3]
                    + pre[(size_t)t * G4H + gate * HID + j];
            gbuf[u * 4 + gate] = g;
        }
        __syncthreads();

        if ((tid & 15) == 0) {  // one leader per unit
            float gi = sigm(gbuf[u * 4 + 0]);
            float gf = sigm(gbuf[u * 4 + 1]);
            float gg = tanhf(gbuf[u * 4 + 2]);
            float go = sigm(gbuf[u * 4 + 3]);
            creg = gf * creg + gi * gg;
            hreg = go * tanhf(creg);
            seq_out[(size_t)t * FEAT + dgrp * HID + j] = hreg;
            h_state[j] = hreg;
        }
        __threadfence();
        grid_barrier(mybar, 64);
    }

    if ((tid & 15) == 0) {
        (hn0 + dgrp * HID)[j] = hreg;
        (cn0 + dgrp * HID)[j] = creg;
    }
}

// ---------------------------------------------------------------------------
// sig_out[t] = sigmoid(dot(layer2_backward_out[t], fc_w[7]) + fc_b[7])
// One wave32 per row, float4 lane-strided loads + shuffle reduction.
// ---------------------------------------------------------------------------
__global__ __launch_bounds__(256) void fc_head(const float* __restrict__ seq,
                                               const float* __restrict__ fcw,
                                               const float* __restrict__ fcb,
                                               float* __restrict__ out) {
    int gid  = blockIdx.x * blockDim.x + threadIdx.x;
    int row  = gid >> 5;
    int lane = threadIdx.x & 31;
    if (row < TSTEPS) {
        const float* hrow = seq + (size_t)row * FEAT + HID;  // backward half
        const float* w = fcw + 7 * HID;
        float acc = 0.0f;
#pragma unroll
        for (int k = lane * 4; k < HID; k += 128) {
            float4 hv = *(const float4*)(hrow + k);
            float4 wv = *(const float4*)(w + k);
            acc = fmaf(hv.x, wv.x, acc);
            acc = fmaf(hv.y, wv.y, acc);
            acc = fmaf(hv.z, wv.z, acc);
            acc = fmaf(hv.w, wv.w, acc);
        }
#pragma unroll
        for (int off = 16; off > 0; off >>= 1) acc += __shfl_down(acc, off, 32);
        if (lane == 0) out[row] = sigm(acc + fcb[7]);
    }
}

// ---------------------------------------------------------------------------
// Host orchestration
// ---------------------------------------------------------------------------
extern "C" void kernel_launch(void* const* d_in, const int* in_sizes, int n_in,
                              void* d_out, int out_size, void* d_ws, size_t ws_size,
                              hipStream_t stream) {
    (void)in_sizes; (void)n_in; (void)out_size; (void)ws_size;

    const float* x    = (const float*)d_in[0];
    const float* w_ih = (const float*)d_in[3];   // (3,2,4096,2048)
    const float* w_hh = (const float*)d_in[4];   // (3,2,4096,1024)
    const float* b_ih = (const float*)d_in[5];   // (3,2,4096)
    const float* b_hh = (const float*)d_in[6];
    const float* fc_w = (const float*)d_in[7];   // (8,1024)
    const float* fc_b = (const float*)d_in[8];   // (8,)
    float* out = (float*)d_out;

    // workspace carve-up
    char* ws = (char*)d_ws;
    const size_t SEQ_BF = (size_t)TSTEPS * FEAT * 2;   // 16 MiB
    const size_t WIH_BF = (size_t)G4H * FEAT * 2;      // 16 MiB
    const size_t WHH_BF = (size_t)G4H * HID * 2;       //  8 MiB
    const size_t PRE_B  = (size_t)TSTEPS * G4H * 4;    // 64 MiB
    const size_t SEQO_B = (size_t)TSTEPS * FEAT * 4;   // 32 MiB

    size_t off = 256;
    unsigned int*   bar     = (unsigned int*)ws;
    unsigned short* seq_in  = (unsigned short*)(ws + off); off += SEQ_BF;
    unsigned short* wih_bf  = (unsigned short*)(ws + off); off += WIH_BF;
    unsigned short* whh0    = (unsigned short*)(ws + off); off += WHH_BF;
    unsigned short* whh1    = (unsigned short*)(ws + off); off += WHH_BF;
    float*          pre0    = (float*)(ws + off); off += PRE_B;
    float*          pre1    = (float*)(ws + off); off += PRE_B;
    float*          seq_out = (float*)(ws + off); off += SEQO_B;
    float*          h0g     = (float*)(ws + off); off += (size_t)HID * 4;
    float*          h1g     = (float*)(ws + off);

    zero_bar<<<1, 32, 0, stream>>>(bar);

    // layer-0 input: x (T x 2048) -> bf16
    {
        int n4 = (TSTEPS * FEAT) / 4;
        f32_to_bf16_vec<<<(n4 + 255) / 256, 256, 0, stream>>>(x, (unsigned int*)seq_in, n4);
    }

    for (int l = 0; l < NUM_LAYERS; ++l) {
        // both directions' pre-activations via WMMA GEMM
        for (int d = 0; d < 2; ++d) {
            const size_t ld = (size_t)(l * 2 + d);
            int n4 = (G4H * FEAT) / 4;
            f32_to_bf16_vec<<<(n4 + 255) / 256, 256, 0, stream>>>(
                w_ih + ld * G4H * FEAT, (unsigned int*)wih_bf, n4);
            dim3 grid(TSTEPS / 128, G4H / 64);
            gemm_preact<<<grid, 256, 0, stream>>>(seq_in, wih_bf,
                                                  b_ih + ld * G4H,
                                                  b_hh + ld * G4H,
                                                  d ? pre1 : pre0);
        }
        // recurrent weights for both directions
        {
            int n4 = (G4H * HID) / 4;
            f32_to_bf16_vec<<<(n4 + 255) / 256, 256, 0, stream>>>(
                w_hh + (size_t)(l * 2 + 0) * G4H * HID, (unsigned int*)whh0, n4);
            f32_to_bf16_vec<<<(n4 + 255) / 256, 256, 0, stream>>>(
                w_hh + (size_t)(l * 2 + 1) * G4H * HID, (unsigned int*)whh1, n4);
        }
        // fused fwd+bwd persistent recurrence (independent barrier domains)
        lstm_recur_bidir<<<128, 256, 0, stream>>>(
            pre0, pre1, whh0, whh1, h0g, h1g, seq_out,
            out + TSTEPS + (size_t)(2 * l) * HID,
            out + TSTEPS + 6 * HID + (size_t)(2 * l) * HID,
            bar);

        if (l + 1 < NUM_LAYERS) {
            int n4 = (TSTEPS * FEAT) / 4;
            f32_to_bf16_vec<<<(n4 + 255) / 256, 256, 0, stream>>>(
                seq_out, (unsigned int*)seq_in, n4);
        }
    }

    fc_head<<<(TSTEPS * 32 + 255) / 256, 256, 0, stream>>>(seq_out, fc_w, fc_b, out);
}
